// TemplatePairStack_64707977281624
// MI455X (gfx1250) — matrix operand order repaired and checked by reading, not compiled, not run on hardware
//
#include <hip/hip_runtime.h>
#include <hip/hip_bf16.h>

typedef _Float16 v16h __attribute__((ext_vector_type(16)));
typedef float    v8f  __attribute__((ext_vector_type(8)));

#define SS 4
#define NN 192
#define CT 64
static const long MTOT = (long)SS * NN * NN;     // 147456 rows
static const long MC   = MTOT * CT;              // 9437184 elements

// ---------------------------------------------------------------------------
// WMMA helpers (CDNA5 gfx1250, wave32): D = A(16x32 f16) x B(32x16 f16) + C(f32)
// ---------------------------------------------------------------------------
__device__ inline v8f wmma_f16(v16h a, v16h b, v8f c) {
  return __builtin_amdgcn_wmma_f32_16x16x32_f16(false, a, false, b, (short)0, c,
                                                false, false);
}

// A fragment: 16x32 f16, row-major, contiguous k order (not ISA table order):
// lanes 0-15 hold rows 0-15 with K in {0..15}; lanes 16-31 K in {16..31}
__device__ inline v16h load_a_frag(const _Float16* A, int lda) {
  int lane = threadIdx.x & 31;
  int m = lane & 15, kg = (lane >> 4) * 16;
  const _Float16* row = A + m * lda;
  v16h a;
#pragma unroll
  for (int e = 0; e < 8; ++e) { a[e] = row[kg + e]; a[8 + e] = row[kg + 8 + e]; }
  return a;
}

// B fragment: 32x16 f16, B row-major [K][N]: lane n holds column n; lanes 0-15 K=0..15,
// lanes 16-31 K=16..31.
__device__ inline v16h load_b_frag(const _Float16* B, int ldb) {
  int lane = threadIdx.x & 31;
  int n = lane & 15, k0 = (lane >> 4) * 16;
  v16h b;
#pragma unroll
  for (int e = 0; e < 16; ++e) b[e] = B[(k0 + e) * ldb + n];
  return b;
}

// B^T fragment from row-major Bt[N][K] (contiguous per lane)
__device__ inline v16h load_bt_frag(const _Float16* Bt, int ldb) {
  int lane = threadIdx.x & 31;
  int n = lane & 15, k0 = (lane >> 4) * 16;
  const _Float16* row = Bt + n * ldb + k0;
  v16h b;
#pragma unroll
  for (int e = 0; e < 16; ++e) b[e] = row[e];
  return b;
}

// ---------------------------------------------------------------------------
// LayerNorm over C=64: one wave per row, 2 elements per lane
// ---------------------------------------------------------------------------
template <bool OUT16>
__global__ __launch_bounds__(256) void ln_kernel(const float* __restrict__ in,
                                                 const float* __restrict__ g,
                                                 const float* __restrict__ b,
                                                 _Float16* __restrict__ outh,
                                                 float* __restrict__ outf, long Mrows) {
  long row = (long)blockIdx.x * 8 + (threadIdx.x >> 5);
  int lane = threadIdx.x & 31;
  if (row >= Mrows) return;
  const float* x = in + row * 64;
  float v0 = x[lane], v1 = x[lane + 32];
  float s = v0 + v1;
#pragma unroll
  for (int off = 16; off; off >>= 1) s += __shfl_xor(s, off, 32);
  float mean = s * (1.f / 64.f);
  float d0 = v0 - mean, d1 = v1 - mean;
  float q = d0 * d0 + d1 * d1;
#pragma unroll
  for (int off = 16; off; off >>= 1) q += __shfl_xor(q, off, 32);
  float rstd = rsqrtf(q * (1.f / 64.f) + 1e-5f);
  float y0 = d0 * rstd * g[lane] + b[lane];
  float y1 = d1 * rstd * g[lane + 32] + b[lane + 32];
  if constexpr (OUT16) {
    outh[row * 64 + lane] = (_Float16)y0;
    outh[row * 64 + lane + 32] = (_Float16)y1;
  } else {
    outf[row * 64 + lane] = y0;
    outf[row * 64 + lane + 32] = y1;
  }
}

// ---------------------------------------------------------------------------
// Generic WMMA GEMM: C[M,NC] = A[M,K](f16) x B[K,NC](f32 weights -> f16) + epilogue
// 8 waves/block, each wave = one 16-row tile of a 128-row slab.
// ---------------------------------------------------------------------------
enum { EPI_F16 = 0, EPI_F16_SCALE, EPI_SIGMOID, EPI_RELU, EPI_RESID, EPI_RESID_GATE,
       EPI_RESID_MASK };

template <int NC, int K, int EPI>
__global__ __launch_bounds__(256) void gemm_kernel(
    const _Float16* __restrict__ A, const float* __restrict__ B,
    const float* __restrict__ bias, float scale, const _Float16* __restrict__ gate,
    const float* __restrict__ maskrow, _Float16* __restrict__ outh,
    float* __restrict__ resid) {
  __shared__ alignas(16) _Float16 Asl[128 * K];
  __shared__ alignas(16) _Float16 Bsl[K * NC];
  int t = threadIdx.x;
  long m0 = (long)blockIdx.x * 128;
  const uint4* srcA = (const uint4*)(A + m0 * K);
  uint4* dstA = (uint4*)Asl;
  for (int i = t; i < 128 * K / 8; i += 256) dstA[i] = srcA[i];
  for (int i = t; i < K * NC; i += 256) Bsl[i] = (_Float16)B[i];
  __syncthreads();
  int w = t >> 5, lane = t & 31, n = lane & 15, hi = lane >> 4;
  for (int nt = 0; nt < NC / 16; ++nt) {
    v8f acc = {};
#pragma unroll
    for (int ks = 0; ks < K / 32; ++ks) {
      v16h a = load_a_frag(Asl + w * 16 * K + ks * 32, K);
      v16h b = load_b_frag(Bsl + ks * 32 * NC + nt * 16, NC);
      acc = wmma_f16(a, b, acc);
    }
    int col = nt * 16 + n;
    long mrow0 = m0 + w * 16 + hi * 8;
#pragma unroll
    for (int r = 0; r < 8; ++r) {
      long m = mrow0 + r;
      float v = acc[r];
      if constexpr (EPI == EPI_F16_SCALE) v *= scale;
      if (bias) v += bias[col];
      if constexpr (EPI == EPI_SIGMOID) v = 1.f / (1.f + __expf(-v));
      if constexpr (EPI == EPI_RELU) v = fmaxf(v, 0.f);
      if constexpr (EPI == EPI_F16 || EPI == EPI_F16_SCALE || EPI == EPI_SIGMOID ||
                    EPI == EPI_RELU) {
        outh[m * NC + col] = (_Float16)v;
      } else if constexpr (EPI == EPI_RESID) {
        resid[m * 64 + col] += v;
      } else if constexpr (EPI == EPI_RESID_GATE) {
        resid[m * 64 + col] += v * (float)gate[m * 64 + col];
      } else {  // EPI_RESID_MASK
        resid[m * 64 + col] += v * maskrow[m];
      }
    }
  }
}

// ---------------------------------------------------------------------------
// Triangle bias: tb[s,h,q,k] = sum_c x[s,q,k,c] * w_b[c,h]
// ---------------------------------------------------------------------------
__global__ __launch_bounds__(256) void tb_kernel(const _Float16* __restrict__ xh,
                                                 const float* __restrict__ wb,
                                                 float* __restrict__ tb) {
  long m = (long)blockIdx.x * 256 + threadIdx.x;
  long s = m / (NN * NN), r = m % (NN * NN), q = r / NN, k = r % NN;
  float acc[4] = {0.f, 0.f, 0.f, 0.f};
  const _Float16* x = xh + m * 64;
  for (int c = 0; c < 64; ++c) {
    float xv = (float)x[c];
#pragma unroll
    for (int h = 0; h < 4; ++h) acc[h] += xv * wb[c * 4 + h];
  }
#pragma unroll
  for (int h = 0; h < 4; ++h) tb[((s * 4 + h) * NN + q) * NN + k] = acc[h];
}

// ---------------------------------------------------------------------------
// Fused triangle attention for one (s,i,h): pad D 16->32 so QK^T is one f16 WMMA
// ---------------------------------------------------------------------------
__global__ __launch_bounds__(256) void attn_kernel(
    const _Float16* __restrict__ qh, const _Float16* __restrict__ kh,
    const _Float16* __restrict__ vh, const _Float16* __restrict__ gh,
    const float* __restrict__ tb, const float* __restrict__ mask,
    _Float16* __restrict__ oh) {
  __shared__ _Float16 Kp[192 * 32];
  __shared__ _Float16 Vb[192 * 16];
  __shared__ _Float16 Qp[32 * 32];
  __shared__ float Lg[32 * 196];
  __shared__ _Float16 Pb[32 * 192];
  int h = blockIdx.x, i = blockIdx.y, s = blockIdx.z;
  int t = threadIdx.x;
  long rowbase = ((long)(s * NN + i)) * NN;
  for (int idx = t; idx < 192 * 16; idx += 256) {
    int d = idx & 15, k = idx >> 4;
    Kp[k * 32 + d] = kh[(rowbase + k) * 64 + h * 16 + d];
    Kp[k * 32 + 16 + d] = (_Float16)0.f;
    Vb[k * 16 + d] = vh[(rowbase + k) * 64 + h * 16 + d];
  }
  __syncthreads();
  int w = t >> 5, lane = t & 31, n = lane & 15, hi = lane >> 4;
  const float* tbbase = tb + ((long)(s * 4 + h) * NN) * NN;
  const float* mrow = mask + rowbase;
  for (int q0 = 0; q0 < NN; q0 += 32) {
    for (int idx = t; idx < 32 * 16; idx += 256) {
      int d = idx & 15, qq = idx >> 4;
      Qp[qq * 32 + d] = qh[(rowbase + q0 + qq) * 64 + h * 16 + d];
      Qp[qq * 32 + 16 + d] = (_Float16)0.f;
    }
    __syncthreads();
    for (int tt = w; tt < 24; tt += 8) {  // 2 q-tiles x 12 k-tiles
      int qt = tt / 12, kt = tt % 12;
      v16h a = load_a_frag(Qp + qt * 16 * 32, 32);
      v16h b = load_bt_frag(Kp + kt * 16 * 32, 32);
      v8f acc = {};
      acc = wmma_f16(a, b, acc);
#pragma unroll
      for (int r = 0; r < 8; ++r) {
        int qr = qt * 16 + hi * 8 + r;
        int kk = kt * 16 + n;
        Lg[qr * 196 + kk] =
            acc[r] + (mrow[kk] - 1.f) * 1e9f + tbbase[(long)(q0 + qr) * NN + kk];
      }
    }
    __syncthreads();
    for (int rr = 0; rr < 4; ++rr) {  // softmax, wave per row
      int qr = w * 4 + rr;
      float vals[6], mx = -1e30f;
#pragma unroll
      for (int cb = 0; cb < 6; ++cb) {
        vals[cb] = Lg[qr * 196 + lane + cb * 32];
        mx = fmaxf(mx, vals[cb]);
      }
#pragma unroll
      for (int off = 16; off; off >>= 1) mx = fmaxf(mx, __shfl_xor(mx, off, 32));
      float sum = 0.f;
#pragma unroll
      for (int cb = 0; cb < 6; ++cb) { vals[cb] = __expf(vals[cb] - mx); sum += vals[cb]; }
#pragma unroll
      for (int off = 16; off; off >>= 1) sum += __shfl_xor(sum, off, 32);
      float inv = 1.f / sum;
#pragma unroll
      for (int cb = 0; cb < 6; ++cb)
        Pb[qr * 192 + lane + cb * 32] = (_Float16)(vals[cb] * inv);
    }
    __syncthreads();
    if (w < 2) {  // A·V, one wave per 16-row q tile
      int qt = w;
      v8f acc = {};
#pragma unroll
      for (int ks = 0; ks < 6; ++ks) {
        v16h a = load_a_frag(Pb + qt * 16 * 192 + ks * 32, 192);
        v16h b = load_b_frag(Vb + ks * 32 * 16, 16);
        acc = wmma_f16(a, b, acc);
      }
#pragma unroll
      for (int r = 0; r < 8; ++r) {
        int qr = q0 + qt * 16 + hi * 8 + r;
        long off = (rowbase + qr) * 64 + h * 16 + n;
        oh[off] = (_Float16)(acc[r] * (float)gh[off]);
      }
    }
    __syncthreads();
  }
}

// ---------------------------------------------------------------------------
// Triangle mult a/b projection: sigmoid(x@Wg+bg)*(x@Wp+bp)*mask, scattered into
// [s,c,i,k] (MODE 0, outgoing) or [s,c,k,i] (MODE 1, incoming) f16 layout.
// ---------------------------------------------------------------------------
template <int MODE>
__global__ __launch_bounds__(256) void trimul_ab(
    const _Float16* __restrict__ xh, const float* __restrict__ wg,
    const float* __restrict__ bg, const float* __restrict__ wp,
    const float* __restrict__ bp, const float* __restrict__ mask,
    _Float16* __restrict__ dst) {
  __shared__ alignas(16) _Float16 Asl[128 * 64];
  __shared__ alignas(16) _Float16 Wg[64 * 64];
  __shared__ alignas(16) _Float16 Wp[64 * 64];
  int t = threadIdx.x;
  long m0 = (long)blockIdx.x * 128;
  const uint4* srcA = (const uint4*)(xh + m0 * 64);
  uint4* dA = (uint4*)Asl;
  for (int i = t; i < 1024; i += 256) dA[i] = srcA[i];
  for (int i = t; i < 4096; i += 256) { Wg[i] = (_Float16)wg[i]; Wp[i] = (_Float16)wp[i]; }
  __syncthreads();
  int w = t >> 5, lane = t & 31, n = lane & 15, hi = lane >> 4;
  for (int nt = 0; nt < 4; ++nt) {
    v8f ag = {}, ap = {};
#pragma unroll
    for (int ks = 0; ks < 2; ++ks) {
      v16h a = load_a_frag(Asl + w * 16 * 64 + ks * 32, 64);
      v16h b1 = load_b_frag(Wg + ks * 32 * 64 + nt * 16, 64);
      v16h b2 = load_b_frag(Wp + ks * 32 * 64 + nt * 16, 64);
      ag = wmma_f16(a, b1, ag);
      ap = wmma_f16(a, b2, ap);
    }
    int c = nt * 16 + n;
    long mrow0 = m0 + w * 16 + hi * 8;
#pragma unroll
    for (int r = 0; r < 8; ++r) {
      long m = mrow0 + r;
      float gv = 1.f / (1.f + __expf(-(ag[r] + bg[c])));
      float v = gv * (ap[r] + bp[c]) * mask[m];
      long s = m / (NN * NN), rem = m % (NN * NN), I = rem / NN, K = rem % NN;
      long addr = (MODE == 0) ? ((s * 64 + c) * NN + I) * NN + K
                              : ((s * 64 + c) * NN + K) * NN + I;
      dst[addr] = (_Float16)v;
    }
  }
}

// Batched T[s,c] = ar[s,c] @ br[s,c]^T (192x192x192, K-steps of 32)
__global__ __launch_bounds__(128) void trimul_gemm(const _Float16* __restrict__ ar,
                                                   const _Float16* __restrict__ br,
                                                   float* __restrict__ tbuf) {
  int w = threadIdx.x >> 5;
  int it = blockIdx.x * 4 + w;  // 0..11
  int jt = blockIdx.y;          // 0..11
  int sc = blockIdx.z;          // s*64 + c
  long base = (long)sc * NN * NN;
  const _Float16* Ab = ar + base + it * 16 * NN;
  const _Float16* Bb = br + base + jt * 16 * NN;
  v8f acc = {};
#pragma unroll
  for (int ks = 0; ks < 6; ++ks) {
    v16h a = load_a_frag(Ab + ks * 32, NN);
    v16h b = load_bt_frag(Bb + ks * 32, NN);
    acc = wmma_f16(a, b, acc);
  }
  int lane = threadIdx.x & 31, n = lane & 15, hi = lane >> 4;
  int s = sc >> 6, c = sc & 63;
  int j = jt * 16 + n;
#pragma unroll
  for (int r = 0; r < 8; ++r) {
    int i = it * 16 + hi * 8 + r;
    tbuf[((long)(s * NN + i) * NN + j) * 64 + c] = acc[r];
  }
}

// ---------------------------------------------------------------------------
// Transposes (swap i<->j), coalesced over c
// ---------------------------------------------------------------------------
__global__ __launch_bounds__(256) void transpose_z(const float* __restrict__ in,
                                                   float* __restrict__ out) {
  int i = blockIdx.x, s = blockIdx.y, t = threadIdx.x;
  int c = t & 63;
  for (int j = t >> 6; j < NN; j += 4)
    out[((long)(s * NN + j) * NN + i) * 64 + c] = in[((long)(s * NN + i) * NN + j) * 64 + c];
}

__global__ __launch_bounds__(256) void transpose_mask(const float* __restrict__ in,
                                                      float* __restrict__ out) {
  long idx = (long)blockIdx.x * 256 + threadIdx.x;
  long s = idx / (NN * NN), r = idx % (NN * NN), i = r / NN, j = r % NN;
  out[(s * NN + j) * NN + i] = in[idx];
}

// ---------------------------------------------------------------------------
// Host-side orchestration
// ---------------------------------------------------------------------------
struct AttP { const float *b_g, *b_o, *ln_b, *ln_g, *w_b, *w_g, *w_k, *w_o, *w_q, *w_v; };
struct MulP {
  const float *b_ag, *b_ap, *b_bg, *b_bp, *b_g, *b_z, *ln_in_b, *ln_in_g, *ln_out_b,
      *ln_out_g, *w_ag, *w_ap, *w_bg, *w_bp, *w_g, *w_z;
};
struct PtP { const float *b1, *b2, *ln_b, *ln_g, *w1, *w2; };

static void run_attention(const AttP& p, float* zb, const float* mask, _Float16* xh,
                          _Float16* qh, _Float16* kh, _Float16* vh, _Float16* gh,
                          _Float16* oh, float* tbb, hipStream_t st) {
  ln_kernel<true><<<MTOT / 8, 256, 0, st>>>(zb, p.ln_g, p.ln_b, xh, nullptr, MTOT);
  tb_kernel<<<MTOT / 256, 256, 0, st>>>(xh, p.w_b, tbb);
  gemm_kernel<64, 64, EPI_F16_SCALE><<<MTOT / 128, 256, 0, st>>>(
      xh, p.w_q, nullptr, 0.25f, nullptr, nullptr, qh, nullptr);
  gemm_kernel<64, 64, EPI_F16><<<MTOT / 128, 256, 0, st>>>(
      xh, p.w_k, nullptr, 1.f, nullptr, nullptr, kh, nullptr);
  gemm_kernel<64, 64, EPI_F16><<<MTOT / 128, 256, 0, st>>>(
      xh, p.w_v, nullptr, 1.f, nullptr, nullptr, vh, nullptr);
  gemm_kernel<64, 64, EPI_SIGMOID><<<MTOT / 128, 256, 0, st>>>(
      xh, p.w_g, p.b_g, 1.f, nullptr, nullptr, gh, nullptr);
  attn_kernel<<<dim3(4, NN, SS), 256, 0, st>>>(qh, kh, vh, gh, tbb, mask, oh);
  gemm_kernel<64, 64, EPI_RESID><<<MTOT / 128, 256, 0, st>>>(
      oh, p.w_o, p.b_o, 1.f, nullptr, nullptr, nullptr, zb);
}

static void run_trimul(const MulP& p, int mode, float* zb, const float* mask,
                       _Float16* xh, _Float16* ar, _Float16* br, _Float16* gh,
                       _Float16* th, float* tbuf, hipStream_t st) {
  ln_kernel<true><<<MTOT / 8, 256, 0, st>>>(zb, p.ln_in_g, p.ln_in_b, xh, nullptr, MTOT);
  if (mode == 0) {
    trimul_ab<0><<<MTOT / 128, 256, 0, st>>>(xh, p.w_ag, p.b_ag, p.w_ap, p.b_ap, mask, ar);
    trimul_ab<0><<<MTOT / 128, 256, 0, st>>>(xh, p.w_bg, p.b_bg, p.w_bp, p.b_bp, mask, br);
  } else {
    trimul_ab<1><<<MTOT / 128, 256, 0, st>>>(xh, p.w_ag, p.b_ag, p.w_ap, p.b_ap, mask, ar);
    trimul_ab<1><<<MTOT / 128, 256, 0, st>>>(xh, p.w_bg, p.b_bg, p.w_bp, p.b_bp, mask, br);
  }
  trimul_gemm<<<dim3(3, 12, SS * 64), 128, 0, st>>>(ar, br, tbuf);
  ln_kernel<true><<<MTOT / 8, 256, 0, st>>>(tbuf, p.ln_out_g, p.ln_out_b, th, nullptr, MTOT);
  gemm_kernel<64, 64, EPI_SIGMOID><<<MTOT / 128, 256, 0, st>>>(
      xh, p.w_g, p.b_g, 1.f, nullptr, nullptr, gh, nullptr);
  gemm_kernel<64, 64, EPI_RESID_GATE><<<MTOT / 128, 256, 0, st>>>(
      th, p.w_z, p.b_z, 1.f, gh, nullptr, nullptr, zb);
}

extern "C" void kernel_launch(void* const* d_in, const int* in_sizes, int n_in,
                              void* d_out, int out_size, void* d_ws, size_t ws_size,
                              hipStream_t stream) {
  const float* z_in = (const float*)d_in[0];
  const float* mask = (const float*)d_in[1];
  // jax pytree flatten order (dict keys sorted): blocks[0..1] then ln_b, ln_g.
  // Block keys: att_e, att_s, mul_i, mul_o, pt (alphabetical).
  auto F = [&](int i) { return (const float*)d_in[i]; };
  auto mkAtt = [&](int b0) {
    AttP p;
    p.b_g = F(b0 + 0); p.b_o = F(b0 + 1); p.ln_b = F(b0 + 2); p.ln_g = F(b0 + 3);
    p.w_b = F(b0 + 4); p.w_g = F(b0 + 5); p.w_k = F(b0 + 6); p.w_o = F(b0 + 7);
    p.w_q = F(b0 + 8); p.w_v = F(b0 + 9);
    return p;
  };
  auto mkMul = [&](int b0) {
    MulP p;
    p.b_ag = F(b0 + 0); p.b_ap = F(b0 + 1); p.b_bg = F(b0 + 2); p.b_bp = F(b0 + 3);
    p.b_g = F(b0 + 4); p.b_z = F(b0 + 5); p.ln_in_b = F(b0 + 6); p.ln_in_g = F(b0 + 7);
    p.ln_out_b = F(b0 + 8); p.ln_out_g = F(b0 + 9); p.w_ag = F(b0 + 10);
    p.w_ap = F(b0 + 11); p.w_bg = F(b0 + 12); p.w_bp = F(b0 + 13); p.w_g = F(b0 + 14);
    p.w_z = F(b0 + 15);
    return p;
  };
  auto mkPt = [&](int b0) {
    PtP p;
    p.b1 = F(b0 + 0); p.b2 = F(b0 + 1); p.ln_b = F(b0 + 2); p.ln_g = F(b0 + 3);
    p.w1 = F(b0 + 4); p.w2 = F(b0 + 5);
    return p;
  };

  // workspace carving (256B aligned)
  char* w = (char*)d_ws;
  auto alloc = [&](size_t bytes) {
    char* p = w;
    w += (bytes + 255) & ~(size_t)255;
    return p;
  };
  float* zt = (float*)alloc(MC * 4);
  _Float16* xh = (_Float16*)alloc(MC * 2);
  _Float16* qh = (_Float16*)alloc(MC * 2);   // aliases: ar
  _Float16* kh = (_Float16*)alloc(MC * 2);   // aliases: br
  _Float16* vh = (_Float16*)alloc(MC * 2);
  _Float16* gh = (_Float16*)alloc(MC * 2);
  _Float16* bufE = (_Float16*)alloc(MTOT * 128 * 2);  // oh / th / h1
  float* tbb = (float*)alloc((size_t)SS * 4 * NN * NN * 4);
  float* mt = (float*)alloc((size_t)SS * NN * NN * 4);
  float* tbuf = (float*)alloc(MC * 4);

  float* zb = (float*)d_out;  // residual stream lives in d_out
  hipMemcpyAsync(zb, z_in, MC * sizeof(float), hipMemcpyDeviceToDevice, stream);
  transpose_mask<<<SS * NN * NN / 256, 256, 0, stream>>>(mask, mt);

  for (int b = 0; b < 2; ++b) {
    int base = 2 + 58 * b;
    AttP att_e = mkAtt(base + 0);
    AttP att_s = mkAtt(base + 10);
    MulP mul_i = mkMul(base + 20);
    MulP mul_o = mkMul(base + 36);
    PtP pt = mkPt(base + 52);

    // 1) triangle attention, starting node
    run_attention(att_s, zb, mask, xh, qh, kh, vh, gh, bufE, tbb, stream);
    // 2) triangle attention, ending node: transpose, attend, transpose back
    transpose_z<<<dim3(NN, SS), 256, 0, stream>>>(zb, zt);
    run_attention(att_e, zt, mt, xh, qh, kh, vh, gh, bufE, tbb, stream);
    transpose_z<<<dim3(NN, SS), 256, 0, stream>>>(zt, zb);
    // 3) triangle mult outgoing / 4) incoming
    run_trimul(mul_o, 0, zb, mask, xh, qh, kh, gh, bufE, tbuf, stream);
    run_trimul(mul_i, 1, zb, mask, xh, qh, kh, gh, bufE, tbuf, stream);
    // 5) pair transition
    ln_kernel<true><<<MTOT / 8, 256, 0, stream>>>(zb, pt.ln_g, pt.ln_b, xh, nullptr, MTOT);
    gemm_kernel<128, 64, EPI_RELU><<<MTOT / 128, 256, 0, stream>>>(
        xh, pt.w1, pt.b1, 1.f, nullptr, nullptr, bufE, nullptr);
    gemm_kernel<64, 128, EPI_RESID_MASK><<<MTOT / 128, 256, 0, stream>>>(
        bufE, pt.w2, pt.b2, 1.f, nullptr, mask, nullptr, zb);
  }
  // final LayerNorm (in place on d_out)
  ln_kernel<false><<<MTOT / 8, 256, 0, stream>>>(zb, F(119), F(118), nullptr, zb, MTOT);
  (void)in_sizes; (void)n_in; (void)out_size; (void)ws_size;
}
